// VectorQuantizer_43224550867374
// MI455X (gfx1250) — matrix operand order repaired and checked by reading, not compiled, not run on hardware
//
#include <hip/hip_runtime.h>
#include <hip/hip_bf16.h>
#include <math.h>

typedef __bf16 v16bf __attribute__((ext_vector_type(16)));
typedef __bf16 v8bf  __attribute__((ext_vector_type(8)));
typedef __bf16 v4bf  __attribute__((ext_vector_type(4)));
typedef float  v8f   __attribute__((ext_vector_type(8)));

#define KCODES 8192
#define DIM    512
#define BROWS  32768
#define ROWB   (DIM*2 + 16)              // 1040B padded bf16 row (16B aligned)
#define NBLK   64                        // codes staged per round
#define BUFB   (NBLK*ROWB)               // 66560B per buffer
#define LDS_BYTES (2*BUFB)               // double-buffered = 133120

// ---------------- ws layout ----------------
#define WS_WNORM  0
#define WS_COUNTS 8192
#define WS_IDX    16384
#define WS_SCAL   49152
#define WS_CB16_F 65536                  // float-index offset of bf16 codebook (8 MB)

// ---------------- output layout (floats) ----------------
#define O_QUANT   0
#define O_IDX     (BROWS*DIM)
#define O_SCAL    (O_IDX + BROWS)
#define O_ECS     (O_SCAL + 4)
#define O_EMASUM  (O_ECS + KCODES)
#define O_CB      (O_EMASUM + KCODES*DIM)

__global__ __launch_bounds__(256) void vq_zero_ws(float* wsf) {
    int i = blockIdx.x * 256 + threadIdx.x;
    if (i < KCODES) wsf[WS_COUNTS + i] = 0.0f;
    if (i < 8)      wsf[WS_SCAL + i]  = 0.0f;
}

__global__ __launch_bounds__(256) void vq_cvt_cb(const float* __restrict__ cb,
                                                 __bf16* __restrict__ out) {
    size_t i4 = (size_t)blockIdx.x * 256 + threadIdx.x;   // over K*D/4
    float4 v = ((const float4*)cb)[i4];
    v4bf o = { (__bf16)v.x, (__bf16)v.y, (__bf16)v.z, (__bf16)v.w };
    ((v4bf*)out)[i4] = o;
}

__global__ __launch_bounds__(256) void vq_wnorm(const float* __restrict__ cb,
                                                float* __restrict__ wnorm) {
    int row  = blockIdx.x * 8 + (threadIdx.x >> 5);
    int lane = threadIdx.x & 31;
    const float4* r = (const float4*)(cb + (size_t)row * DIM);
    float s = 0.0f;
    #pragma unroll
    for (int i = 0; i < 4; ++i) {
        float4 v = r[lane + i * 32];
        s += v.x*v.x + v.y*v.y + v.z*v.z + v.w*v.w;
    }
    #pragma unroll
    for (int m = 16; m >= 1; m >>= 1) s += __shfl_xor(s, m, 32);
    if (lane == 0) wnorm[row] = s;
}

__global__ __launch_bounds__(256) void vq_ema_init(const float* __restrict__ ema,
                                                   float* __restrict__ out) {
    size_t i = (size_t)blockIdx.x * 256 + threadIdx.x;
    out[i] = 0.99f * ema[i];
}

// ---- fused bf16-WMMA distance + argmin ----
// A in VGPRs; async double-buffered B; dual-tile + 1-chunk-ahead pipeline.
__global__ __launch_bounds__(256, 1) void vq_argmin(const float* __restrict__ z,
                                                    const __bf16* __restrict__ cb16,
                                                    const float* __restrict__ wnorm,
                                                    int* __restrict__ idx_out,
                                                    float* __restrict__ idxf_out) {
    extern __shared__ char lds[];
    const int tid   = threadIdx.x;
    const int lane  = tid & 31;
    const int wave  = tid >> 5;
    const int l16   = lane & 15;
    const int lsel  = lane >> 4;
    const int mbase = blockIdx.x * 128;
    const unsigned ldsbase = (unsigned)(uintptr_t)lds;

    // lane's 16 A fragments straight from global f32 (one-time)
    const float* Ar = z + (size_t)(mbase + wave * 16 + l16) * DIM;
    v16bf A[16];
    #pragma unroll
    for (int ck = 0; ck < 16; ++ck) {
        const int d0 = ck * 32;
        float4 lo0 = *(const float4*)(Ar + d0 + lsel * 16);
        float4 lo1 = *(const float4*)(Ar + d0 + lsel * 16 + 4);
        float4 hi0 = *(const float4*)(Ar + d0 + lsel * 16 + 8);
        float4 hi1 = *(const float4*)(Ar + d0 + lsel * 16 + 12);
        v16bf a = { (__bf16)lo0.x,(__bf16)lo0.y,(__bf16)lo0.z,(__bf16)lo0.w,
                    (__bf16)lo1.x,(__bf16)lo1.y,(__bf16)lo1.z,(__bf16)lo1.w,
                    (__bf16)hi0.x,(__bf16)hi0.y,(__bf16)hi0.z,(__bf16)hi0.w,
                    (__bf16)hi1.x,(__bf16)hi1.y,(__bf16)hi1.z,(__bf16)hi1.w };
        A[ck] = a;
    }

    float best[8]; int bestn[8];
    #pragma unroll
    for (int v = 0; v < 8; ++v) { best[v] = 3.4e38f; bestn[v] = 0; }

    const int NR = KCODES / NBLK;        // 128 rounds of 64 codes

    auto stage = [&](int r, int buf) {
        const unsigned dbase = ldsbase + (unsigned)buf * BUFB;
        const char* gbase = (const char*)cb16 + ((size_t)r * NBLK << 10);
        for (int e = tid; e < NBLK * 64; e += 256) {      // 16B chunks
            int row = e >> 6, c16 = e & 63;
            unsigned laddr = dbase + (unsigned)(row * ROWB + c16 * 16);
            const char* g = gbase + ((size_t)row << 10) + (c16 << 4);
            asm volatile("global_load_async_to_lds_b128 %0, %1, off"
                         :: "v"(laddr), "v"(g) : "memory");
        }
    };

    stage(0, 0);
    for (int r = 0; r < NR; ++r) {
        asm volatile("s_wait_asynccnt 0x0" ::: "memory");
        __syncthreads();
        if (r + 1 < NR) stage(r + 1, (r + 1) & 1);

        const char* bufp = lds + (r & 1) * BUFB;
        #pragma unroll
        for (int tp = 0; tp < 2; ++tp) {                  // two tile-pairs per round
            const char* B0 = bufp + (tp * 32 + l16) * ROWB;
            const char* B1 = B0 + 16 * ROWB;
            const int boff = lsel * 32;                   // byte offset of K half

            v8f c0 = {}, c1 = {};
            v8bf c0lo = *(const v8bf*)(B0 + boff);
            v8bf c0hi = *(const v8bf*)(B0 + boff + 16);
            v8bf c1lo = *(const v8bf*)(B1 + boff);
            v8bf c1hi = *(const v8bf*)(B1 + boff + 16);
            #pragma unroll
            for (int ck = 0; ck < 16; ++ck) {
                v8bf n0lo = c0lo, n0hi = c0hi, n1lo = c1lo, n1hi = c1hi;
                if (ck < 15) {                            // prefetch next chunk
                    const int d1 = (ck + 1) * 64 + boff;
                    n0lo = *(const v8bf*)(B0 + d1);
                    n0hi = *(const v8bf*)(B0 + d1 + 16);
                    n1lo = *(const v8bf*)(B1 + d1);
                    n1hi = *(const v8bf*)(B1 + d1 + 16);
                }
                v16bf b0 = __builtin_shufflevector(c0lo, c0hi,
                    0,1,2,3,4,5,6,7,8,9,10,11,12,13,14,15);
                v16bf b1 = __builtin_shufflevector(c1lo, c1hi,
                    0,1,2,3,4,5,6,7,8,9,10,11,12,13,14,15);
                c0 = __builtin_amdgcn_wmma_f32_16x16x32_bf16(
                         false, A[ck], false, b0, (short)0, c0, false, false);
                c1 = __builtin_amdgcn_wmma_f32_16x16x32_bf16(
                         false, A[ck], false, b1, (short)0, c1, false, false);
                c0lo = n0lo; c0hi = n0hi; c1lo = n1lo; c1hi = n1hi;
            }
            const int na = r * NBLK + tp * 32 + l16;
            const int nb = na + 16;
            const float wna = wnorm[na], wnb = wnorm[nb];
            #pragma unroll
            for (int v = 0; v < 8; ++v) {                 // score = ||w||^2 - 2 z.w
                float s0 = wna - 2.0f * c0[v];
                if (s0 < best[v]) { best[v] = s0; bestn[v] = na; }
                float s1 = wnb - 2.0f * c1[v];
                if (s1 < best[v]) { best[v] = s1; bestn[v] = nb; }
            }
        }
    }

    // argmin reduce across the 16 lanes of each half (rows v / v+8)
    #pragma unroll
    for (int v = 0; v < 8; ++v) {
        float s = best[v]; int n = bestn[v];
        #pragma unroll
        for (int m = 8; m >= 1; m >>= 1) {
            float s2 = __shfl_xor(s, m, 32);
            int   n2 = __shfl_xor(n, m, 32);
            if (s2 < s || (s2 == s && n2 < n)) { s = s2; n = n2; }
        }
        if (l16 == 0) {
            int row = mbase + wave * 16 + v + lsel * 8;
            idx_out[row]  = n;
            idxf_out[row] = (float)n;
        }
    }
}

// gather quantized, loss partials, EMA scatter-adds. One wave per z row.
__global__ __launch_bounds__(256) void vq_gather(const float* __restrict__ z,
                                                 const float* __restrict__ cb,
                                                 const int* __restrict__ idx,
                                                 float* __restrict__ outq,
                                                 float* __restrict__ emasum,
                                                 float* __restrict__ counts,
                                                 float* __restrict__ loss_acc) {
    int b    = blockIdx.x * 8 + (threadIdx.x >> 5);
    int lane = threadIdx.x & 31;
    int k    = idx[b];
    const float4* zr = (const float4*)(z  + (size_t)b * DIM);
    const float4* cr = (const float4*)(cb + (size_t)k * DIM);
    float4*       qo = (float4*)(outq + (size_t)b * DIM);
    float*        es = emasum + (size_t)k * DIM;
    float s = 0.0f;
    #pragma unroll
    for (int i = 0; i < 4; ++i) {
        int d4 = lane + i * 32;
        float4 zv = zr[d4], qv = cr[d4];
        float4 o; o.x = zv.x + (qv.x - zv.x); o.y = zv.y + (qv.y - zv.y);
                  o.z = zv.z + (qv.z - zv.z); o.w = zv.w + (qv.w - zv.w);
        qo[d4] = o;
        float dx = zv.x - qv.x, dy = zv.y - qv.y, dz = zv.z - qv.z, dw = zv.w - qv.w;
        s += dx*dx + dy*dy + dz*dz + dw*dw;
        atomicAdd(es + d4 * 4 + 0, 0.01f * zv.x);
        atomicAdd(es + d4 * 4 + 1, 0.01f * zv.y);
        atomicAdd(es + d4 * 4 + 2, 0.01f * zv.z);
        atomicAdd(es + d4 * 4 + 3, 0.01f * zv.w);
    }
    #pragma unroll
    for (int m = 16; m >= 1; m >>= 1) s += __shfl_xor(s, m, 32);
    if (lane == 0) {
        atomicAdd(loss_acc, s);
        atomicAdd(&counts[k], 1.0f);
    }
}

__global__ __launch_bounds__(256) void vq_ecs(const float* __restrict__ ema_cs,
                                              const float* __restrict__ counts,
                                              float* __restrict__ out_ecs,
                                              float* __restrict__ n_acc,
                                              float* __restrict__ ent_acc) {
    int k    = blockIdx.x * 256 + threadIdx.x;
    int lane = threadIdx.x & 31;
    float cs = counts[k];
    float ne = 0.99f * ema_cs[k] + 0.01f * cs;
    out_ecs[k] = ne;
    float p = cs / (float)BROWS;
    float e = p * logf(p + 1e-10f);
    #pragma unroll
    for (int m = 16; m >= 1; m >>= 1) { ne += __shfl_xor(ne, m, 32); e += __shfl_xor(e, m, 32); }
    if (lane == 0) { atomicAdd(n_acc, ne); atomicAdd(ent_acc, e); }
}

__global__ __launch_bounds__(256) void vq_cbout(const float* __restrict__ out_ecs,
                                                const float* __restrict__ n_acc,
                                                const float* __restrict__ emasum,
                                                float* __restrict__ out_cb) {
    size_t i = (size_t)blockIdx.x * 256 + threadIdx.x;
    int k = (int)(i >> 9);
    float n = *n_acc;
    float csn = (out_ecs[k] + 1e-5f) / (n + (float)KCODES * 1e-5f) * n;
    out_cb[i] = emasum[i] / csn;
}

__global__ void vq_scalars(const float* loss_acc, const float* ent_acc, float* out) {
    if (threadIdx.x == 0) {
        float mean = loss_acc[0] / ((float)BROWS * (float)DIM);
        out[0] = mean;
        out[1] = mean;
        out[2] = mean + 0.25f * mean;
        out[3] = expf(-ent_acc[0]);
    }
}

extern "C" void kernel_launch(void* const* d_in, const int* in_sizes, int n_in,
                              void* d_out, int out_size, void* d_ws, size_t ws_size,
                              hipStream_t stream) {
    const float* z       = (const float*)d_in[0];
    const float* cb      = (const float*)d_in[1];
    const float* ema_cs  = (const float*)d_in[2];
    const float* ema_sum = (const float*)d_in[3];
    float* out = (float*)d_out;
    float* wsf = (float*)d_ws;

    float*  wnorm    = wsf + WS_WNORM;
    float*  counts   = wsf + WS_COUNTS;
    int*    idx      = (int*)(wsf + WS_IDX);
    float*  loss_acc = wsf + WS_SCAL + 0;
    float*  n_acc    = wsf + WS_SCAL + 1;
    float*  ent_acc  = wsf + WS_SCAL + 2;
    __bf16* cb16     = (__bf16*)(wsf + WS_CB16_F);

    (void)hipFuncSetAttribute((const void*)vq_argmin,
        hipFuncAttributeMaxDynamicSharedMemorySize, LDS_BYTES);

    vq_zero_ws<<<(KCODES + 255) / 256, 256, 0, stream>>>(wsf);
    vq_cvt_cb<<<(KCODES * DIM / 4) / 256, 256, 0, stream>>>(cb, cb16);
    vq_wnorm<<<KCODES / 8, 256, 0, stream>>>(cb, wnorm);
    vq_ema_init<<<(KCODES * DIM) / 256, 256, 0, stream>>>(ema_sum, out + O_EMASUM);
    vq_argmin<<<BROWS / 128, 256, LDS_BYTES, stream>>>(z, cb16, wnorm, idx, out + O_IDX);
    vq_gather<<<BROWS / 8, 256, 0, stream>>>(z, cb, idx, out + O_QUANT,
                                             out + O_EMASUM, counts, loss_acc);
    vq_ecs<<<KCODES / 256, 256, 0, stream>>>(ema_cs, counts, out + O_ECS, n_acc, ent_acc);
    vq_cbout<<<(KCODES * DIM) / 256, 256, 0, stream>>>(out + O_ECS, n_acc,
                                                       out + O_EMASUM, out + O_CB);
    vq_scalars<<<1, 32, 0, stream>>>(loss_acc, ent_acc, out + O_SCAL);
}
